// SoftmaxMatcherBlock_16269336118052
// MI455X (gfx1250) — compile-verified, matrix operand order checked by reading, not run on hardware
//
#include <hip/hip_runtime.h>
#include <hip/hip_bf16.h>
#include <math.h>

// ---------------- problem constants ----------------
constexpr int  Bn = 2;
constexpr int  Cc = 256;            // channels
constexpr int  Nn = 1024;           // src points
constexpr int  Mm = 65536;          // H*W target pixels
constexpr int  G  = 8;              // grid split over M (partials reduced later)
constexpr int  MPG = Mm / G;        // 8192 m per group
constexpr int  NT = 32;             // n rows per workgroup
constexpr int  MT = 64;             // m cols per LDS tile
constexpr int  KSTR  = 264;         // K1 [m][c] row stride (bf16), rows 16B aligned
constexpr int  K2STR = 72;          // K2 [c][m] row stride (bf16), rows 16B aligned
constexpr int  QSTR = 256;          // Qlds [n][c]
constexpr int  PSTR = 72;           // Plds [n][m]
constexpr int  ROWS = 262;          // partials: 0=den,1..3=coords,4=w,5..260=desc,261=pmax

constexpr float CART_MIN = (256.0f / 2.0f - 0.5f) * 0.2592f;   // 33.048
constexpr float INV_RES  = 1.0f / 0.2592f;

// ---------------- output float offsets (concatenated return order) ----------
constexpr size_t OUT_COORDS = 0;                                // B x 3 x N
constexpr size_t OUT_W      = OUT_COORDS + (size_t)Bn * 3 * Nn; // B x 1 x N
constexpr size_t OUT_DESC   = OUT_W      + (size_t)Bn * Nn;     // B x C x N
constexpr size_t OUT_2D     = OUT_DESC   + (size_t)Bn * Cc * Nn;// B x N x 2
constexpr size_t OUT_VALID  = OUT_2D     + (size_t)Bn * Nn * 2; // B x 1 x N
constexpr size_t OUT_MAX    = OUT_VALID  + (size_t)Bn * Nn;     // B

// ---------------- workspace byte offsets (~19 MB total) ----------------
constexpr size_t WSB_INV  = 0;                                   // B*M f32 logit scales
constexpr size_t WSB_Q    = WSB_INV  + (size_t)Bn * Mm * 4;      // B*N*C bf16 packed Q
constexpr size_t WSB_PART = WSB_Q    + (size_t)Bn * Nn * Cc * 2; // B*G*ROWS*N f32 partials
constexpr size_t WSB_DTOT = WSB_PART + (size_t)Bn * G * ROWS * Nn * 4; // B*N f32

// ---------------- dynamic LDS carve (98 KB < 320 KB/WGP on CDNA5) ---------
constexpr size_t LDS_K1   = 0;                                    // [m][c] bf16
constexpr size_t LDS_K2   = LDS_K1  + (size_t)MT * KSTR * 2;      // [c][m] bf16
constexpr size_t LDS_Q    = LDS_K2  + (size_t)Cc * K2STR * 2;     // [n][c] bf16
constexpr size_t LDS_P    = LDS_Q   + (size_t)NT * QSTR * 2;      // [n][m] bf16
constexpr size_t LDS_AUX  = LDS_P   + (size_t)NT * PSTR * 2;      // 4 x MT f32
constexpr size_t LDS_INV  = LDS_AUX + (size_t)4 * MT * 4;         // MT f32
constexpr size_t LDS_DRED = LDS_INV + (size_t)MT * 4;             // 4 x NT f32
constexpr size_t LDS_DMAX = LDS_DRED + (size_t)4 * NT * 4;        // 4 x NT f32
constexpr size_t LDS_ARED = LDS_DMAX + (size_t)4 * NT * 4;        // 8 x NT x 4 f32
constexpr size_t LDS_BYTES = LDS_ARED + (size_t)8 * NT * 4 * 4;   // = 98048

typedef __attribute__((ext_vector_type(16))) __bf16 v16bf;
typedef __attribute__((ext_vector_type(8)))  __bf16 bf16x8;
typedef __attribute__((ext_vector_type(4)))  __bf16 bf16x4;
typedef __attribute__((ext_vector_type(8)))  float  v8f;

__device__ __forceinline__ size_t PART(int b, int g, int row, int n) {
  return ((size_t)(b * G + g) * ROWS + row) * Nn + n;
}

// ============================================================================
// Kernel 1: per-column stats of tgt_desc.
// zn over C: since src_desc_norm columns are zero-mean, the GEMM only needs
// 1/std per column; fold in the /C and the 1/temp=100 softmax scale.
// ============================================================================
__global__ __launch_bounds__(256) void matcher_stats(
    const float* __restrict__ tgt_desc, float* __restrict__ ws_inv) {
  const size_t t = (size_t)blockIdx.x * 256 + threadIdx.x;   // B*M threads
  const int    b = (int)(t / Mm);
  const size_t m = t % Mm;
  const float* p = tgt_desc + (size_t)b * Cc * Mm + m;
  float s = 0.f, ss = 0.f;
  #pragma unroll 4
  for (int c = 0; c < Cc; ++c) {
    float x = p[(size_t)c * Mm];    // coalesced across threads (consecutive m)
    s += x; ss += x * x;
  }
  const float mean = s * (1.0f / Cc);
  const float var  = (ss - (float)Cc * mean * mean) * (1.0f / (Cc - 1));
  const float sd   = sqrtf(fmaxf(var, 1e-30f));
  ws_inv[t] = 100.0f / (sd * (float)Cc);   // logit = dot * ws_inv
}

// ============================================================================
// Kernel 2: pack src_desc_norm (B,C,N) -> bf16 Q (B,N,C) for WMMA A operands.
// ============================================================================
__global__ __launch_bounds__(256) void matcher_packq(
    const float* __restrict__ src, __bf16* __restrict__ q) {
  const int t = blockIdx.x * 256 + threadIdx.x;     // B*N*C
  const int c = t % Cc;
  const int n = (t / Cc) % Nn;
  const int b = t / (Cc * Nn);
  q[t] = (__bf16)src[((size_t)(b * Cc + c)) * Nn + n];
}

// ============================================================================
// Kernel 3: fused flash-style matcher.
// Per workgroup (ntile, mgroup, b): stream M in 64-wide tiles. Each tile is
// staged ONCE from HBM (float4 loads) and kept in LDS in BOTH layouts:
//   K1 [m][c]  -> GEMM1 B operand (k=c contiguous, ds_load_b128)
//   K2 [c][m]  -> GEMM2 B operand (k=m contiguous, ds_load_b128)
// so every WMMA fragment load is two 16B LDS vector loads (no scalar gather).
// Fixed-shift softmax (ZNCC logit <= 100) removes flash-style rescaling;
// row denom / row pmax / coord sums are accumulated alongside.
// 8 waves: GEMM1 split (nhalf x msub), GEMM2 split (nhalf x cgroup).
// ============================================================================
__global__ __launch_bounds__(256) void matcher_main(
    const float* __restrict__ tgt_desc,
    const float* __restrict__ tgt_coords,
    const float* __restrict__ tgt_w,
    const float* __restrict__ ws_inv,
    const __bf16* __restrict__ qpack,
    float* __restrict__ part) {
  extern __shared__ __align__(16) char smem[];
  __bf16* K1     = (__bf16*)(smem + LDS_K1);
  __bf16* K2     = (__bf16*)(smem + LDS_K2);
  __bf16* Qlds   = (__bf16*)(smem + LDS_Q);
  __bf16* Plds   = (__bf16*)(smem + LDS_P);
  float*  auxlds = (float*)(smem + LDS_AUX);    // [4][MT]
  float*  invlds = (float*)(smem + LDS_INV);    // [MT]
  float*  dred   = (float*)(smem + LDS_DRED);   // [4][NT]
  float*  dmaxr  = (float*)(smem + LDS_DMAX);   // [4][NT]
  float*  auxred = (float*)(smem + LDS_ARED);   // [8][NT][4]

  const int nt   = blockIdx.x;         // n tile (fast dim -> L2 reuse of K slice)
  const int g    = blockIdx.y;         // m group
  const int b    = blockIdx.z;
  const int tid  = threadIdx.x;
  const int lane = tid & 31;
  const int wave = tid >> 5;
  const int n0   = nt * NT;
  const int mg0  = g * MPG;

  // stage Q strip (32 x 256 bf16) once
  for (int i = tid; i < NT * Cc; i += 256) {
    const int n = i >> 8, c = i & 255;
    Qlds[n * QSTR + c] = qpack[((size_t)(b * Nn + n0 + n)) * Cc + c];
  }

  const int nhalf = wave >> 2;                 // 16-row n strip
  const int msub  = wave & 3;                  // GEMM1: 16-col m substrip
  const int cgrp  = wave & 3;                  // GEMM2: 64-col c group
  const int arow  = nhalf * 16 + (lane & 15);  // A-fragment row (ISA 16-bit A layout)
  const int koff  = (lane >> 4) << 3;          // k offset 0/8 from lane half
  const int auxn  = tid & 31;
  const int auxg  = tid >> 5;

  v8f   acc[4] = {};                           // GEMM2 accum: 16n x 64c
  float dsum[8] = {};
  float pmax[8] = {};
  float aux[4]  = {0.f, 0.f, 0.f, 0.f};

  for (int mb = 0; mb < MPG; mb += MT) {
    const int mbase = mg0 + mb;
    __syncthreads();   // protect K1/K2/Plds from previous iteration's readers

    // ---- stage K tile: fp32 global (float4, coalesced) -> bf16 LDS x2 ----
    for (int i = tid; i < (Cc * MT) / 4; i += 256) {
      const int c = i >> 4;             // 16 float4 per c-row
      const int m = (i & 15) * 4;
      const float4 v =
          *(const float4*)&tgt_desc[((size_t)(b * Cc + c)) * Mm + mbase + m];
      const __bf16 b0 = (__bf16)v.x, b1 = (__bf16)v.y,
                   b2 = (__bf16)v.z, b3 = (__bf16)v.w;
      *(bf16x4*)&K2[c * K2STR + m] = bf16x4{b0, b1, b2, b3};  // [c][m] b64 store
      K1[(m + 0) * KSTR + c] = b0;                            // [m][c] scatter
      K1[(m + 1) * KSTR + c] = b1;
      K1[(m + 2) * KSTR + c] = b2;
      K1[(m + 3) * KSTR + c] = b3;
    }
    if (mb + MT < MPG)    // CDNA5 global_prefetch_b8 of next tile's cache lines
      __builtin_prefetch(&tgt_desc[((size_t)(b * Cc + tid)) * Mm + mbase + MT], 0, 1);
    if (tid < MT) {
      const int m = tid;
      invlds[m]           = ws_inv[(size_t)b * Mm + mbase + m];
      auxlds[0 * MT + m]  = tgt_coords[((size_t)(b * 3 + 0)) * Mm + mbase + m];
      auxlds[1 * MT + m]  = tgt_coords[((size_t)(b * 3 + 1)) * Mm + mbase + m];
      auxlds[2 * MT + m]  = tgt_coords[((size_t)(b * 3 + 2)) * Mm + mbase + m];
      auxlds[3 * MT + m]  = tgt_w[(size_t)b * Mm + mbase + m];
    }
    __syncthreads();

    // ---- GEMM1: logits(16x16) = Q(16x256) x K^T, K=32 bf16 WMMA chain ----
    v8f lg = {};
    const int bmrow = msub * 16 + (lane & 15);   // B operand: n=m over lanes
    #pragma unroll
    for (int kc = 0; kc < Cc; kc += 32) {
      bf16x8 alo = *(const bf16x8*)&Qlds[arow * QSTR + kc + koff];
      bf16x8 ahi = *(const bf16x8*)&Qlds[arow * QSTR + kc + 16 + koff];
      bf16x8 blo = *(const bf16x8*)&K1[bmrow * KSTR + kc + koff];
      bf16x8 bhi = *(const bf16x8*)&K1[bmrow * KSTR + kc + 16 + koff];
      v16bf A, Bv;
      #pragma unroll
      for (int e = 0; e < 8; ++e) { A[e] = alo[e]; A[e + 8] = ahi[e];
                                    Bv[e] = blo[e]; Bv[e + 8] = bhi[e]; }
      lg = __builtin_amdgcn_wmma_f32_16x16x32_bf16(false, A, false, Bv,
                                                   (short)0, lg, false, false);
    }

    // ---- fixed-shift softmax numerators; track denom + row max ----
    const int   mcol = msub * 16 + (lane & 15);
    const float inv  = invlds[mcol];
    #pragma unroll
    for (int gg = 0; gg < 8; ++gg) {
      const float p = __expf(lg[gg] * inv - 100.0f);   // logit <= 100 always
      dsum[gg] += p;
      pmax[gg]  = fmaxf(pmax[gg], p);
      const int nrow = nhalf * 16 + gg + ((lane >> 4) << 3);
      Plds[nrow * PSTR + mcol] = (__bf16)p;
    }
    __syncthreads();

    // ---- GEMM2: O(16n x 64c) += P(16 x 64) @ V(64 x 64c), V == K tile ----
    #pragma unroll
    for (int ks = 0; ks < 2; ++ks) {
      bf16x8 plo = *(const bf16x8*)&Plds[arow * PSTR + ks * 32 + koff];
      bf16x8 phi = *(const bf16x8*)&Plds[arow * PSTR + ks * 32 + 16 + koff];
      v16bf Ap;
      #pragma unroll
      for (int e = 0; e < 8; ++e) { Ap[e] = plo[e]; Ap[e + 8] = phi[e]; }
      #pragma unroll
      for (int cf = 0; cf < 4; ++cf) {
        const int ccol = cgrp * 64 + cf * 16 + (lane & 15);
        // k=m contiguous in K2 [c][m]: two 16B vector LDS loads per fragment
        bf16x8 blo = *(const bf16x8*)&K2[ccol * K2STR + ks * 32 + koff];
        bf16x8 bhi = *(const bf16x8*)&K2[ccol * K2STR + ks * 32 + 16 + koff];
        v16bf Bv;
        #pragma unroll
        for (int e = 0; e < 8; ++e) { Bv[e] = blo[e]; Bv[e + 8] = bhi[e]; }
        acc[cf] = __builtin_amdgcn_wmma_f32_16x16x32_bf16(false, Ap, false, Bv,
                                                (short)0, acc[cf], false, false);
      }
    }

    // ---- coords/weight weighted sums (4 channels, VALU) ----
    #pragma unroll
    for (int mm = 0; mm < 8; ++mm) {
      const int m = auxg * 8 + mm;
      const float p = (float)Plds[auxn * PSTR + m];
      aux[0] += p * auxlds[0 * MT + m];
      aux[1] += p * auxlds[1 * MT + m];
      aux[2] += p * auxlds[2 * MT + m];
      aux[3] += p * auxlds[3 * MT + m];
    }
  }

  // ---- epilogue: fixed-order reductions, write partials for this m-group ----
  #pragma unroll
  for (int gg = 0; gg < 8; ++gg) {
    float v = dsum[gg], mx = pmax[gg];
    v += __shfl_xor(v, 1); v += __shfl_xor(v, 2);
    v += __shfl_xor(v, 4); v += __shfl_xor(v, 8);
    mx = fmaxf(mx, __shfl_xor(mx, 1)); mx = fmaxf(mx, __shfl_xor(mx, 2));
    mx = fmaxf(mx, __shfl_xor(mx, 4)); mx = fmaxf(mx, __shfl_xor(mx, 8));
    if ((lane & 15) == 0) {
      const int nrow = nhalf * 16 + gg + ((lane >> 4) << 3);
      dred[msub * NT + nrow]  = v;
      dmaxr[msub * NT + nrow] = mx;
    }
  }
  auxred[(auxg * NT + auxn) * 4 + 0] = aux[0];
  auxred[(auxg * NT + auxn) * 4 + 1] = aux[1];
  auxred[(auxg * NT + auxn) * 4 + 2] = aux[2];
  auxred[(auxg * NT + auxn) * 4 + 3] = aux[3];
  __syncthreads();

  if (tid < NT) {
    const int n = n0 + tid;
    part[PART(b, g, 0, n)] =
        dred[0 * NT + tid] + dred[1 * NT + tid] + dred[2 * NT + tid] + dred[3 * NT + tid];
    part[PART(b, g, ROWS - 1, n)] =
        fmaxf(fmaxf(dmaxr[0 * NT + tid], dmaxr[1 * NT + tid]),
              fmaxf(dmaxr[2 * NT + tid], dmaxr[3 * NT + tid]));
    float a0 = 0.f, a1 = 0.f, a2 = 0.f, a3 = 0.f;
    #pragma unroll
    for (int k = 0; k < 8; ++k) {
      a0 += auxred[(k * NT + tid) * 4 + 0];
      a1 += auxred[(k * NT + tid) * 4 + 1];
      a2 += auxred[(k * NT + tid) * 4 + 2];
      a3 += auxred[(k * NT + tid) * 4 + 3];
    }
    part[PART(b, g, 1, n)] = a0; part[PART(b, g, 2, n)] = a1;
    part[PART(b, g, 3, n)] = a2; part[PART(b, g, 4, n)] = a3;
  }

  {  // GEMM2 accumulator fragments -> desc partial rows (C/D lane layout)
    const int nbase = n0 + nhalf * 16 + ((lane >> 4) << 3);
    #pragma unroll
    for (int cf = 0; cf < 4; ++cf) {
      const int c = cgrp * 64 + cf * 16 + (lane & 15);
      #pragma unroll
      for (int gg = 0; gg < 8; ++gg)
        part[PART(b, g, 5 + c, nbase + gg)] = acc[cf][gg];
    }
  }
}

// ============================================================================
// Kernel 4: per-(b,n) combine partials, normalize softmax, zn(desc), project.
// ============================================================================
__global__ __launch_bounds__(256) void matcher_finish(
    const float* __restrict__ part, float* __restrict__ dtot,
    float* __restrict__ out) {
  __shared__ float red[256];
  __shared__ float shc[3];
  __shared__ float sh_mean, sh_istd;
  const int n = blockIdx.x;
  const int b = blockIdx.y;
  const int c = threadIdx.x;

  float den = 0.f;
  #pragma unroll
  for (int g = 0; g < G; ++g) den += part[PART(b, g, 0, n)];
  const float invden = 1.0f / den;

  float oc = 0.f;
  #pragma unroll
  for (int g = 0; g < G; ++g) oc += part[PART(b, g, 5 + c, n)];
  oc *= invden;

  red[c] = oc; __syncthreads();
  #pragma unroll
  for (int s = 128; s > 0; s >>= 1) { if (c < s) red[c] += red[c + s]; __syncthreads(); }
  if (c == 0) sh_mean = red[0] * (1.0f / Cc);
  __syncthreads();
  const float d = oc - sh_mean;
  red[c] = d * d; __syncthreads();
  #pragma unroll
  for (int s = 128; s > 0; s >>= 1) { if (c < s) red[c] += red[c + s]; __syncthreads(); }
  if (c == 0) sh_istd = rsqrtf(fmaxf(red[0] * (1.0f / (Cc - 1)), 1e-30f));
  __syncthreads();
  out[OUT_DESC + (size_t)(b * Cc + c) * Nn + n] = d * sh_istd;

  if (c < 3) {
    float v = 0.f;
    #pragma unroll
    for (int g = 0; g < G; ++g) v += part[PART(b, g, 1 + c, n)];
    v *= invden;
    out[OUT_COORDS + (size_t)(b * 3 + c) * Nn + n] = v;
    shc[c] = v;
  }
  if (c == 3) {
    float w = 0.f;
    #pragma unroll
    for (int g = 0; g < G; ++g) w += part[PART(b, g, 4, n)];
    out[OUT_W + (size_t)b * Nn + n] = w * invden;
  }
  if (c == 4) out[OUT_VALID + (size_t)b * Nn + n] = 1.0f;
  if (c == 5) {
    float pm = 0.f;
    #pragma unroll
    for (int g = 0; g < G; ++g) pm = fmaxf(pm, part[PART(b, g, ROWS - 1, n)]);
    dtot[(size_t)b * Nn + n] = pm * invden;   // max softmax prob of row n
  }
  __syncthreads();
  if (c == 0) {
    out[OUT_2D + ((size_t)b * Nn + n) * 2 + 0] = (CART_MIN + shc[1]) * INV_RES;
    out[OUT_2D + ((size_t)b * Nn + n) * 2 + 1] = (CART_MIN - shc[0]) * INV_RES;
  }
}

// ============================================================================
// Kernel 5: max_softmax[b] = max over n of per-row max probability.
// ============================================================================
__global__ __launch_bounds__(256) void matcher_maxsm(
    const float* __restrict__ dtot, float* __restrict__ out) {
  __shared__ float red[256];
  const int b = blockIdx.x, t = threadIdx.x;
  float mx = 0.f;
  for (int n = t; n < Nn; n += 256) mx = fmaxf(mx, dtot[(size_t)b * Nn + n]);
  red[t] = mx; __syncthreads();
  #pragma unroll
  for (int s = 128; s > 0; s >>= 1) { if (t < s) red[t] = fmaxf(red[t], red[t + s]); __syncthreads(); }
  if (t == 0) out[OUT_MAX + b] = red[0];
}

// ============================================================================
extern "C" void kernel_launch(void* const* d_in, const int* in_sizes, int n_in,
                              void* d_out, int out_size, void* d_ws, size_t ws_size,
                              hipStream_t stream) {
  const float* tgt_coords = (const float*)d_in[1];   // B x 3 x M
  const float* tgt_w      = (const float*)d_in[3];   // B x 1 x H x W
  const float* src_desc   = (const float*)d_in[5];   // B x C x N (already zn'd)
  const float* tgt_desc   = (const float*)d_in[6];   // B x C x H x W
  float* out = (float*)d_out;

  char*   ws      = (char*)d_ws;
  float*  ws_inv  = (float*)(ws + WSB_INV);
  __bf16* ws_q    = (__bf16*)(ws + WSB_Q);
  float*  ws_part = (float*)(ws + WSB_PART);
  float*  ws_dtot = (float*)(ws + WSB_DTOT);

  matcher_stats<<<(Bn * Mm) / 256, 256, 0, stream>>>(tgt_desc, ws_inv);
  matcher_packq<<<(Bn * Nn * Cc) / 256, 256, 0, stream>>>(src_desc, ws_q);
  matcher_main<<<dim3(Nn / NT, G, Bn), 256, (int)LDS_BYTES, stream>>>(
      tgt_desc, tgt_coords, tgt_w, ws_inv, ws_q, ws_part);
  matcher_finish<<<dim3(Nn, Bn), 256, 0, stream>>>(ws_part, ws_dtot, out);
  matcher_maxsm<<<Bn, 256, 0, stream>>>(ws_dtot, out);
}
